// Head_21509196218471
// MI455X (gfx1250) — matrix operand order repaired and checked by reading, not compiled, not run on hardware
//
#include <hip/hip_runtime.h>
#include <hip/hip_bf16.h>

// Causal single-head attention for MI455X (gfx1250, wave32, WMMA).
//   K0: one-shot f32->bf16 conversion of x and Wq/Wk/Wv (L2-resident after).
//   K1: Q/K/V projection GEMM, pure-bf16 inner loop (b128 loads + 4x WMMA).
//   K2: flash attention, split-K=4 over key blocks (flash-decoding style).
//   K3: combine split partials (m, l, O) and normalize.

#define B_    4
#define T_    2048
#define C_    1024
#define HS_   64
#define WAVES 4
#define SPLIT 4
#define NQT   (T_ / 16)          // query tiles per batch
#define PREC  (16 + 16 + 16*64)  // floats per partial record: m[16], l[16], O[16][64]

typedef __bf16 bf16_t;
typedef bf16_t v16bf __attribute__((ext_vector_type(16)));
typedef bf16_t v8bf  __attribute__((ext_vector_type(8)));
typedef float  v8f   __attribute__((ext_vector_type(8)));

__device__ __forceinline__ v8f wmma_bf16(v16bf a, v16bf b, v8f c) {
  // D = A(16x32 bf16) * B(32x16 bf16) + C(16x16 f32)
  return __builtin_amdgcn_wmma_f32_16x16x32_bf16(
      /*neg_a=*/false, a, /*neg_b=*/false, b,
      /*c_mod=*/(short)0, c, /*reuse_a=*/false, /*reuse_b=*/false);
}

__device__ __forceinline__ v16bf pack16(v8bf lo, v8bf hi) {
  v16bf r;
#pragma unroll
  for (int i = 0; i < 8; ++i) { r[i] = lo[i]; r[8 + i] = hi[i]; }
  return r;
}

// A-fragment (16x32 bf16, row-major source): lane owns row = lane&15,
// ksel = (lane>>4)*8; elements [2*ksel..2*ksel+15] (contiguous).
__device__ __forceinline__ v16bf load_afrag(const bf16_t* rowp, int ksel) {
  v8bf lo = *(const v8bf*)(rowp + 2 * ksel);
  v8bf hi = *(const v8bf*)(rowp + 2 * ksel + 8);
  return pack16(lo, hi);
}

// B-fragment (32x16 bf16) where this lane's 16 K-elements are contiguous.
__device__ __forceinline__ v16bf load_bfrag_contig(const bf16_t* p) {
  v8bf lo = *(const v8bf*)p;
  v8bf hi = *(const v8bf*)(p + 8);
  return pack16(lo, hi);
}

// ---------------- K0: f32 -> bf16 conversion of x and W ----------------
__global__ __launch_bounds__(256)
void cvt_kernel(const float* __restrict__ x,
                const float* __restrict__ Wq,
                const float* __restrict__ Wk,
                const float* __restrict__ Wv,
                bf16_t* __restrict__ xb, bf16_t* __restrict__ wb) {
  const size_t NX = (size_t)B_ * T_ * C_;
  const size_t NW = (size_t)HS_ * C_;
  const size_t i  = ((size_t)blockIdx.x * blockDim.x + threadIdx.x) * 4;
  if (i < NX) {
    float4 v = *(const float4*)(x + i);
    bf16_t* o = xb + i;
    o[0] = (bf16_t)v.x; o[1] = (bf16_t)v.y; o[2] = (bf16_t)v.z; o[3] = (bf16_t)v.w;
  } else {
    const size_t j = i - NX;
    if (j < 3 * NW) {
      const float* W = (j < NW) ? Wq : (j < 2 * NW) ? Wk : Wv;
      float4 v = *(const float4*)(W + (j % NW));
      bf16_t* o = wb + j;
      o[0] = (bf16_t)v.x; o[1] = (bf16_t)v.y; o[2] = (bf16_t)v.z; o[3] = (bf16_t)v.w;
    }
  }
}

// ---------------- K1: Q/K/V projection GEMM (pure bf16) ----------------
// grid = (B*T/16, 3); one wave computes a 16x64 output tile of one matrix.
__global__ __launch_bounds__(32)
void qkv_gemm_kernel(const bf16_t* __restrict__ xb,
                     const bf16_t* __restrict__ wb,
                     bf16_t* __restrict__ Qb,
                     bf16_t* __restrict__ Kb,
                     bf16_t* __restrict__ Vt) {
  const int lane = threadIdx.x & 31;
  const int tile = blockIdx.x;
  const int mt   = blockIdx.y;  // 0=Q, 1=K, 2=V
  const int row0 = tile * 16;
  const int arow = lane & 15;
  const int hi   = lane >> 4;
  const int ksel = hi * 8;
  const int kb16 = hi * 16;
  const int col  = lane & 15;

  const bf16_t* xrow = xb + (size_t)(row0 + arow) * C_;
  // B[k][n] = W[nt*16+n][k]; this lane's 16 k's are contiguous along C.
  const bf16_t* w0 = wb + (size_t)mt * HS_ * C_ + (size_t)(0 * 16 + col) * C_ + kb16;
  const bf16_t* w1 = w0 + (size_t)16 * C_;
  const bf16_t* w2 = w0 + (size_t)32 * C_;
  const bf16_t* w3 = w0 + (size_t)48 * C_;

  v8f a0 = {}, a1 = {}, a2 = {}, a3 = {};
#pragma unroll 2
  for (int k0 = 0; k0 < C_; k0 += 32) {
    const v16bf a = load_afrag(xrow + k0, ksel);     // reused by 4 WMMAs
    a0 = wmma_bf16(a, load_bfrag_contig(w0 + k0), a0);
    a1 = wmma_bf16(a, load_bfrag_contig(w1 + k0), a1);
    a2 = wmma_bf16(a, load_bfrag_contig(w2 + k0), a2);
    a3 = wmma_bf16(a, load_bfrag_contig(w3 + k0), a3);
  }

  // D layout: acc[r] -> (row0 + r + 8*hi, nt*16 + col)
#pragma unroll
  for (int r = 0; r < 8; ++r) {
    const int gr = row0 + r + 8 * hi;
    const bf16_t v0 = (bf16_t)a0[r], v1 = (bf16_t)a1[r];
    const bf16_t v2 = (bf16_t)a2[r], v3 = (bf16_t)a3[r];
    if (mt == 0) {
      bf16_t* p = Qb + (size_t)gr * HS_ + col;
      p[0] = v0; p[16] = v1; p[32] = v2; p[48] = v3;
    } else if (mt == 1) {
      bf16_t* p = Kb + (size_t)gr * HS_ + col;
      p[0] = v0; p[16] = v1; p[32] = v2; p[48] = v3;
    } else {  // V stored transposed: Vt[b][h][t]
      const int bb = gr / T_, tt = gr % T_;
      bf16_t* p = Vt + ((size_t)bb * HS_ + col) * T_ + tt;
      p[(size_t)0 * T_]  = v0; p[(size_t)16 * T_] = v1;
      p[(size_t)32 * T_] = v2; p[(size_t)48 * T_] = v3;
    }
  }
}

// ---------------- K2: flash attention, split-K over key blocks ----------------
// grid = (T/16/WAVES, SPLIT, B), block = 32*WAVES. Wave -> (query tile, split).
__global__ __launch_bounds__(32 * WAVES)
void attn_partial_kernel(const bf16_t* __restrict__ Qb,
                         const bf16_t* __restrict__ Kb,
                         const bf16_t* __restrict__ Vt,
                         float* __restrict__ part) {
  __shared__ __align__(64) bf16_t pbuf[WAVES][16 * 32];

  const int lane  = threadIdx.x & 31;
  const int wv    = threadIdx.x >> 5;
  const int sp    = blockIdx.y;
  const int b     = blockIdx.z;
  const int qt    = blockIdx.x * WAVES + wv;
  const int qbase = qt * 16;

  const bf16_t* Q = Qb + (size_t)b * T_ * HS_;
  const bf16_t* K = Kb + (size_t)b * T_ * HS_;
  const bf16_t* V = Vt + (size_t)b * HS_ * T_;  // [HS][T]

  const int arow = lane & 15;
  const int hi   = lane >> 4;
  const int ksel = hi * 8;
  const int kb16 = hi * 16;
  const int col  = lane & 15;

  // Q tile 16x64 -> two A fragments (HS contraction split 32+32)
  const bf16_t* qrow = Q + (size_t)(qbase + arow) * HS_;
  const v16bf qa0 = load_afrag(qrow, ksel);
  const v16bf qa1 = load_afrag(qrow + 32, ksel);

  v8f o0 = {}, o1 = {}, o2 = {}, o3 = {};
  float m[8], l[8];
#pragma unroll
  for (int r = 0; r < 8; ++r) { m[r] = -1e30f; l[r] = 0.0f; }

  const float scale = 0.125f;  // 1/sqrt(HS)
  const int nb     = (qbase + 16 + 31) / 32;       // causal key blocks
  const int blkLo  = (sp * nb) / SPLIT;
  const int blkHi  = ((sp + 1) * nb) / SPLIT;
  bf16_t* pb = pbuf[wv];

  for (int blk = blkLo; blk < blkHi; ++blk) {
    const int j0 = blk * 32;
    if (blk + 1 < blkHi) {  // warm caches for next key block
      __builtin_prefetch(K + (size_t)(j0 + 32 + arow) * HS_, 0, 3);
      __builtin_prefetch(V + (size_t)(hi * 32 + col) * T_ + j0 + 32, 0, 3);
    }

    // S = Q*K^T : two 16x16 column tiles, K=64 over HS (4 WMMAs, interleaved)
    v8f s0 = {}, s1 = {};
    {
      const bf16_t* k0p = K + (size_t)(j0 + col) * HS_ + kb16;
      const bf16_t* k1p = K + (size_t)(j0 + 16 + col) * HS_ + kb16;
      s0 = wmma_bf16(qa0, load_bfrag_contig(k0p), s0);
      s1 = wmma_bf16(qa0, load_bfrag_contig(k1p), s1);
      s0 = wmma_bf16(qa1, load_bfrag_contig(k0p + 32), s0);
      s1 = wmma_bf16(qa1, load_bfrag_contig(k1p + 32), s1);
    }
    s0 = s0 * scale;
    s1 = s1 * scale;

    // causal mask (C layout: row = qbase + r + 8*hi, col = j0 + jt*16 + col)
#pragma unroll
    for (int r = 0; r < 8; ++r) {
      const int gr = qbase + r + 8 * hi;
      if (j0 + col > gr)      s0[r] = -1e30f;
      if (j0 + 16 + col > gr) s1[r] = -1e30f;
    }

    // online softmax: row stats reduced across each 16-lane half-wave.
    // NOTE: with split-K a whole block row can be masked while m is still
    // -1e30 -> exp(s-m)=exp(0)=1 would be wrong; guard masked lanes to 0.
    float p0[8], p1[8];
#pragma unroll
    for (int r = 0; r < 8; ++r) {
      float rm = fmaxf(s0[r], s1[r]);
#pragma unroll
      for (int off = 1; off < 16; off <<= 1)
        rm = fmaxf(rm, __shfl_xor(rm, off, 16));
      const float mn    = fmaxf(m[r], rm);
      const float alpha = __expf(m[r] - mn);
      p0[r] = (s0[r] > -1e29f) ? __expf(s0[r] - mn) : 0.0f;
      p1[r] = (s1[r] > -1e29f) ? __expf(s1[r] - mn) : 0.0f;
      float rs = p0[r] + p1[r];
#pragma unroll
      for (int off = 1; off < 16; off <<= 1)
        rs += __shfl_xor(rs, off, 16);
      l[r] = l[r] * alpha + rs;
      m[r] = mn;
      o0[r] *= alpha; o1[r] *= alpha; o2[r] *= alpha; o3[r] *= alpha;
    }

    // C-layout P -> LDS (row-major 16x32 bf16) -> A-fragment reload
#pragma unroll
    for (int r = 0; r < 8; ++r) {
      const int lr = r + 8 * hi;
      pb[lr * 32 + col]      = (bf16_t)p0[r];
      pb[lr * 32 + 16 + col] = (bf16_t)p1[r];
    }
    asm volatile("s_wait_dscnt 0" ::: "memory");  // cross-lane LDS W->R fence
    const v16bf pa = load_afrag(pb + arow * 32, ksel);
    asm volatile("" ::: "memory");

    // O += P*V : V as B fragments from transposed layout (contiguous in T)
    const bf16_t* vbase = V + j0 + kb16;
    o0 = wmma_bf16(pa, load_bfrag_contig(vbase + (size_t)(0 * 16 + col) * T_), o0);
    o1 = wmma_bf16(pa, load_bfrag_contig(vbase + (size_t)(1 * 16 + col) * T_), o1);
    o2 = wmma_bf16(pa, load_bfrag_contig(vbase + (size_t)(2 * 16 + col) * T_), o2);
    o3 = wmma_bf16(pa, load_bfrag_contig(vbase + (size_t)(3 * 16 + col) * T_), o3);
  }

  // write partial record: m[16], l[16], O[16][64] (unnormalized)
  float* pr = part + (size_t)(((b * NQT + qt) * SPLIT) + sp) * PREC;
#pragma unroll
  for (int r = 0; r < 8; ++r) {
    const int lr = r + 8 * hi;
    pr[lr]      = m[r];   // 16 lanes store the same replicated value
    pr[16 + lr] = l[r];
    float* po = pr + 32 + (size_t)lr * HS_ + col;
    po[0] = o0[r]; po[16] = o1[r]; po[32] = o2[r]; po[48] = o3[r];
  }
}

// ---------------- K3: combine split partials ----------------
// grid = (NQT, B), block = 256; each thread handles 4 of the 16x64 outputs.
__global__ __launch_bounds__(256)
void combine_kernel(const float* __restrict__ part, float* __restrict__ out) {
  const int qt = blockIdx.x;
  const int b  = blockIdx.y;
  const float* pbase = part + (size_t)((b * NQT + qt) * SPLIT) * PREC;
  for (int e = threadIdx.x; e < 16 * HS_; e += 256) {
    const int row = e >> 6;
    float mt = -1e30f;
#pragma unroll
    for (int s = 0; s < SPLIT; ++s)
      mt = fmaxf(mt, pbase[(size_t)s * PREC + row]);
    float den = 0.0f, acc = 0.0f;
#pragma unroll
    for (int s = 0; s < SPLIT; ++s) {
      const float w = __expf(pbase[(size_t)s * PREC + row] - mt);  // 0 for empty splits
      den += pbase[(size_t)s * PREC + 16 + row] * w;
      acc += pbase[(size_t)s * PREC + 32 + e] * w;
    }
    out[((size_t)b * T_ + qt * 16 + row) * HS_ + (e & 63)] = acc / den;
  }
}

extern "C" void kernel_launch(void* const* d_in, const int* in_sizes, int n_in,
                              void* d_out, int out_size, void* d_ws, size_t ws_size,
                              hipStream_t stream) {
  (void)in_sizes; (void)n_in; (void)out_size; (void)ws_size;
  const float* x  = (const float*)d_in[0];
  const float* Wq = (const float*)d_in[1];
  const float* Wk = (const float*)d_in[2];
  const float* Wv = (const float*)d_in[3];

  // Workspace layout (all 256B aligned):
  //   xb   [B*T][C]    bf16  16 MB
  //   wb   [3][HS][C]  bf16  384 KB
  //   Qb,Kb[B*T][HS]   bf16  1 MB each
  //   Vt   [B][HS][T]  bf16  1 MB
  //   part [B][NQT][SPLIT][PREC] f32  ~8.7 MB
  char* ws = (char*)d_ws;
  constexpr size_t NXB   = (size_t)B_ * T_ * C_ * sizeof(bf16_t);
  constexpr size_t NWB   = (size_t)3 * HS_ * C_ * sizeof(bf16_t);
  constexpr size_t NQKV  = (size_t)B_ * T_ * HS_ * sizeof(bf16_t);
  bf16_t* xb   = (bf16_t*)(ws);
  bf16_t* wb   = (bf16_t*)(ws + NXB);
  bf16_t* qb   = (bf16_t*)(ws + NXB + NWB);
  bf16_t* kb   = (bf16_t*)(ws + NXB + NWB + NQKV);
  bf16_t* vt   = (bf16_t*)(ws + NXB + NWB + 2 * NQKV);
  float*  part = (float*)(ws + NXB + NWB + 3 * NQKV);

  // K0: convert x + W to bf16 (4 elems/thread)
  const size_t nquads = ((size_t)B_ * T_ * C_ + (size_t)3 * HS_ * C_) / 4;
  cvt_kernel<<<dim3((unsigned)((nquads + 255) / 256)), 256, 0, stream>>>(
      x, Wq, Wk, Wv, xb, wb);

  // K1: Q/K/V projection GEMM
  dim3 g1((B_ * T_) / 16, 3);
  qkv_gemm_kernel<<<g1, 32, 0, stream>>>(xb, wb, qb, kb, vt);

  // K2: flash attention partials (split-K)
  dim3 g2(NQT / WAVES, SPLIT, B_);
  attn_partial_kernel<<<g2, 32 * WAVES, 0, stream>>>(qb, kb, vt, part);

  // K3: combine + normalize
  dim3 g3(NQT, B_);
  combine_kernel<<<g3, 256, 0, stream>>>(part, (float*)d_out);
}
